// GCN_51788715655651
// MI455X (gfx1250) — matrix operand order repaired and checked by reading, not compiled, not run on hardware
//
#include <hip/hip_runtime.h>
#include <stdint.h>

// ---------------------------------------------------------------------------
// GCN forward for gfx1250 (MI455X):
//   3x [ bf16 WMMA GEMM -> f32 atomic edge scatter-add -> bias/relu ]
// hw (N x 256 f32 = 100MB) fits in the 192MB L2, so edge gathers/atomics are
// L2-resident. GEMM: one wave computes a 16x64 strip of C (4x v_wmma per
// k-step, A fragment reused 4x), K/M templated so every fragment load is
// base+immediate, fully unrolled, with a 1-step software pipeline so WMMAs
// overlap the next k-step's global_load_b128s instead of waiting loadcnt 0.
// ---------------------------------------------------------------------------

typedef __attribute__((ext_vector_type(16))) __bf16 v16bf;
typedef __attribute__((ext_vector_type(8)))  float  v8f;

union Frag { unsigned int u[8]; v16bf v; };

__device__ __forceinline__ unsigned short f2bf(float f) {
  unsigned int u = __float_as_uint(f);
  u += 0x7fffu + ((u >> 16) & 1u);      // round-to-nearest-even
  return (unsigned short)(u >> 16);
}

// ---- degree count (col occurrences of E edges) ----------------------------
__global__ void k_count(const int* __restrict__ col, int* __restrict__ cnt, int E) {
  int e = blockIdx.x * blockDim.x + threadIdx.x;
  if (e < E) atomicAdd(&cnt[col[e]], 1);
}

// ---- dinv = rsqrt(deg), deg = cnt + 1 (self loop) -------------------------
__global__ void k_dinv(const int* __restrict__ cnt, float* __restrict__ dinv, int n) {
  int i = blockIdx.x * blockDim.x + threadIdx.x;
  if (i < n) dinv[i] = rsqrtf((float)(cnt[i] + 1));
}

// ---- flat f32 -> bf16 conversion ------------------------------------------
__global__ void k_cvt(const float* __restrict__ src, unsigned short* __restrict__ dst,
                      long long count) {
  long long i = (long long)blockIdx.x * blockDim.x + threadIdx.x;
  if (i < count) dst[i] = f2bf(src[i]);
}

// ---- W [K x M] f32 -> Wt [M x K] bf16 (transposed, so B k-pairs are dwords)
__global__ void k_cvt_wt(const float* __restrict__ W, unsigned short* __restrict__ Wt,
                         int K, int M) {
  int i = blockIdx.x * blockDim.x + threadIdx.x;
  if (i < K * M) {
    int k = i / M, m = i % M;
    Wt[(size_t)m * K + k] = f2bf(W[i]);
  }
}

// ---- fragment loads (all offsets compile-time immediates after unroll) ----
__device__ __forceinline__ void load_frags(const unsigned int* __restrict__ Ar,
                                           const unsigned int* __restrict__ Br0,
                                           const unsigned int* __restrict__ Br1,
                                           const unsigned int* __restrict__ Br2,
                                           const unsigned int* __restrict__ Br3,
                                           int kd, Frag& a, Frag& b0, Frag& b1,
                                           Frag& b2, Frag& b3) {
#pragma unroll
  for (int j = 0; j < 8; ++j) {
    // A dword index: (half*16 + 2j + k0)/2, half folded in base
    a.u[j]  = Ar[j + kd];
    // B dword index: (half*16 + 2j + k0)/2, half folded in base
    b0.u[j] = Br0[j + kd];
    b1.u[j] = Br1[j + kd];
    b2.u[j] = Br2[j + kd];
    b3.u[j] = Br3[j + kd];
  }
}

// ---- WMMA GEMM: C[n x M] = A[n x K](bf16) * Wt[M x K]^T(bf16) -------------
// One wave computes a 16x64 strip (4 adjacent 16x16 tiles).
// Fragment layouts as loaded here (wave32, same contiguous k order for A and B):
//   A 16x32 bf16: lanes 0-15 row=lm, VGPR j holds K pair half*16+2j
//   B 32x16 bf16: lane group half covers K=half*16..+15, VGPR j holds K pair 2j
//   C 16x16 f32 : VGPR j -> row half*8+j, col lm
template <int K, int M>
__global__ void k_gemm_t(const unsigned short* __restrict__ A,
                         const unsigned short* __restrict__ Bt,
                         float* __restrict__ C, int n_rows) {
  int wave = (int)((blockIdx.x * (long long)blockDim.x + threadIdx.x) >> 5);
  int lane = threadIdx.x & 31;
  const int tiles_m = n_rows >> 4;     // n_rows divisible by 16 (100000 = 6250*16)
  const int nblks   = M >> 6;          // 64-wide N blocks (M = 256 or 128)
  if (wave >= tiles_m * nblks) return; // wave-uniform: EXEC stays all-ones
  int tile_m = wave / nblks;
  int nblk   = wave - tile_m * nblks;
  int half = lane >> 4;
  int lm   = lane & 15;

  // Base pointers with the lane-half offset pre-folded (dword units).
  const unsigned int* Ar =
      (const unsigned int*)(A + (size_t)(tile_m * 16 + lm) * K) + half * 8;
  const unsigned int* Br0 =
      (const unsigned int*)(Bt + (size_t)(nblk * 64 +  0 + lm) * K) + half * 8;
  const unsigned int* Br1 =
      (const unsigned int*)(Bt + (size_t)(nblk * 64 + 16 + lm) * K) + half * 8;
  const unsigned int* Br2 =
      (const unsigned int*)(Bt + (size_t)(nblk * 64 + 32 + lm) * K) + half * 8;
  const unsigned int* Br3 =
      (const unsigned int*)(Bt + (size_t)(nblk * 64 + 48 + lm) * K) + half * 8;

  v8f acc0 = {}, acc1 = {}, acc2 = {}, acc3 = {};
  constexpr int STEPS = K / 32;        // k-step = 32, dword stride 16

  Frag a, b0, b1, b2, b3;
  load_frags(Ar, Br0, Br1, Br2, Br3, 0, a, b0, b1, b2, b3);

#pragma unroll
  for (int s = 0; s < STEPS; ++s) {
    Frag an, b0n, b1n, b2n, b3n;
    if (s + 1 < STEPS)                 // prefetch next k-step before computing
      load_frags(Ar, Br0, Br1, Br2, Br3, (s + 1) * 16, an, b0n, b1n, b2n, b3n);

    acc0 = __builtin_amdgcn_wmma_f32_16x16x32_bf16(false, a.v, false, b0.v,
                                                   (short)0, acc0, false, false);
    acc1 = __builtin_amdgcn_wmma_f32_16x16x32_bf16(false, a.v, false, b1.v,
                                                   (short)0, acc1, false, false);
    acc2 = __builtin_amdgcn_wmma_f32_16x16x32_bf16(false, a.v, false, b2.v,
                                                   (short)0, acc2, false, false);
    acc3 = __builtin_amdgcn_wmma_f32_16x16x32_bf16(false, a.v, false, b3.v,
                                                   (short)0, acc3, false, false);

    if (s + 1 < STEPS) {               // SSA rename only (loop fully unrolled)
      a = an; b0 = b0n; b1 = b1n; b2 = b2n; b3 = b3n;
    }
  }

  int crow = tile_m * 16 + half * 8;
  float* Crow = C + (size_t)crow * M + nblk * 64 + lm;
#pragma unroll
  for (int j = 0; j < 8; ++j) {
    Crow[(size_t)j * M +  0] = acc0[j];
    Crow[(size_t)j * M + 16] = acc1[j];
    Crow[(size_t)j * M + 32] = acc2[j];
    Crow[(size_t)j * M + 48] = acc3[j];
  }
}

// ---- edge scatter: agg[col] += hw[row] * dinv[row]*dinv[col] --------------
// One wave per edge; lanes cover the feature dim in float4 chunks
// (D is a multiple of 128 -> 32 lanes * 4 floats per pass).
__global__ void k_agg(const int* __restrict__ row, const int* __restrict__ col,
                      const float* __restrict__ dinv, const float* __restrict__ hw,
                      float* __restrict__ agg, int E, int D) {
  int wave = (int)((blockIdx.x * (long long)blockDim.x + threadIdx.x) >> 5);
  int lane = threadIdx.x & 31;
  if (wave >= E) return;
  int r = row[wave], c = col[wave];
  float nrm = dinv[r] * dinv[c];
  const float* src = hw  + (size_t)r * D;
  float*       dst = agg + (size_t)c * D;
  for (int d0 = lane * 4; d0 < D; d0 += 128) {
    float4 v = *(const float4*)(src + d0);           // global_load_b128
    unsafeAtomicAdd(&dst[d0 + 0], v.x * nrm);        // global_atomic_add_f32
    unsafeAtomicAdd(&dst[d0 + 1], v.y * nrm);
    unsafeAtomicAdd(&dst[d0 + 2], v.z * nrm);
    unsafeAtomicAdd(&dst[d0 + 3], v.w * nrm);
  }
}

// ---- finalize (hidden layers): + self-loop + bias, relu, emit bf16 --------
__global__ void k_fin_bf16(const float* __restrict__ agg, const float* __restrict__ hw,
                           const float* __restrict__ dinv, const float* __restrict__ bias,
                           unsigned short* __restrict__ outb, int n, int D) {
  long long idx = (long long)blockIdx.x * blockDim.x + threadIdx.x;
  if (idx >= (long long)n * D) return;
  int i = (int)(idx / D), d = (int)(idx - (long long)i * D);
  float di = dinv[i];
  float v  = agg[idx] + hw[idx] * di * di + bias[d];
  outb[idx] = f2bf(fmaxf(v, 0.0f));
}

// ---- finalize (output layer): + self-loop + bias, f32 out -----------------
__global__ void k_fin_f32(const float* __restrict__ agg, const float* __restrict__ hw,
                          const float* __restrict__ dinv, const float* __restrict__ bias,
                          float* __restrict__ out, int n, int D) {
  long long idx = (long long)blockIdx.x * blockDim.x + threadIdx.x;
  if (idx >= (long long)n * D) return;
  int i = (int)(idx / D), d = (int)(idx - (long long)i * D);
  float di = dinv[i];
  out[idx] = agg[idx] + hw[idx] * di * di + bias[d];
}

static inline void launch_gemm(const unsigned short* A, const unsigned short* Bt,
                               float* C, int n, int K, int M, hipStream_t s) {
  long long waves   = (long long)(n / 16) * (M / 64);
  long long threads = waves * 32;
  int blocks = (int)((threads + 255) / 256);
  if (K == 128 && M == 256)      k_gemm_t<128, 256><<<blocks, 256, 0, s>>>(A, Bt, C, n);
  else if (K == 256 && M == 256) k_gemm_t<256, 256><<<blocks, 256, 0, s>>>(A, Bt, C, n);
  else                           k_gemm_t<256, 128><<<blocks, 256, 0, s>>>(A, Bt, C, n);
}

extern "C" void kernel_launch(void* const* d_in, const int* in_sizes, int n_in,
                              void* d_out, int out_size, void* d_ws, size_t ws_size,
                              hipStream_t stream) {
  const int DIN = 128, DH = 256, DOUT = 128;
  const float* x   = (const float*)d_in[0];
  const int*   ei  = (const int*)d_in[1];
  const float* W0  = (const float*)d_in[4];
  const float* b0  = (const float*)d_in[5];
  const float* W1  = (const float*)d_in[6];
  const float* b1  = (const float*)d_in[7];
  const float* W2  = (const float*)d_in[8];
  const float* b2  = (const float*)d_in[9];

  const int N = in_sizes[0] / DIN;
  const int E = in_sizes[1] / 2;
  const int* rowp = ei;       // edge_index[0]
  const int* colp = ei + E;   // edge_index[1]

  // ---- carve workspace (~257 MB) ----
  char*  ws  = (char*)d_ws;
  size_t off = 0;
  auto carve = [&](size_t bytes) -> void* {
    void* p = ws + off;
    off += (bytes + 255) & ~(size_t)255;
    return p;
  };
  int*            cnt  = (int*)carve((size_t)N * 4);
  float*          dinv = (float*)carve((size_t)N * 4);
  unsigned short* Abf  = (unsigned short*)carve((size_t)N * DH * 2);
  float*          hw   = (float*)carve((size_t)N * DH * 4);
  float*          agg  = (float*)carve((size_t)N * DH * 4);
  unsigned short* Wt0  = (unsigned short*)carve((size_t)DIN * DH * 2);
  unsigned short* Wt1  = (unsigned short*)carve((size_t)DH * DH * 2);
  unsigned short* Wt2  = (unsigned short*)carve((size_t)DH * DOUT * 2);

  const int TB = 256;

  // ---- normalization ----
  hipMemsetAsync(cnt, 0, (size_t)N * 4, stream);
  k_count<<<(E + TB - 1) / TB, TB, 0, stream>>>(colp, cnt, E);
  k_dinv<<<(N + TB - 1) / TB, TB, 0, stream>>>(cnt, dinv, N);

  // ---- precision conversion ----
  k_cvt<<<(int)(((long long)N * DIN + TB - 1) / TB), TB, 0, stream>>>(x, Abf, (long long)N * DIN);
  k_cvt_wt<<<(DIN * DH  + TB - 1) / TB, TB, 0, stream>>>(W0, Wt0, DIN, DH);
  k_cvt_wt<<<(DH  * DH  + TB - 1) / TB, TB, 0, stream>>>(W1, Wt1, DH, DH);
  k_cvt_wt<<<(DH  * DOUT + TB - 1) / TB, TB, 0, stream>>>(W2, Wt2, DH, DOUT);

  int aggBlocks = (int)(((long long)E * 32 + TB - 1) / TB);

  // ---- layer 0: 128 -> 256, relu ----
  launch_gemm(Abf, Wt0, hw, N, DIN, DH, stream);
  hipMemsetAsync(agg, 0, (size_t)N * DH * 4, stream);
  k_agg<<<aggBlocks, TB, 0, stream>>>(rowp, colp, dinv, hw, agg, E, DH);
  k_fin_bf16<<<(int)(((long long)N * DH + TB - 1) / TB), TB, 0, stream>>>(agg, hw, dinv, b0, Abf, N, DH);

  // ---- layer 1: 256 -> 256, relu ----
  launch_gemm(Abf, Wt1, hw, N, DH, DH, stream);
  hipMemsetAsync(agg, 0, (size_t)N * DH * 4, stream);
  k_agg<<<aggBlocks, TB, 0, stream>>>(rowp, colp, dinv, hw, agg, E, DH);
  k_fin_bf16<<<(int)(((long long)N * DH + TB - 1) / TB), TB, 0, stream>>>(agg, hw, dinv, b1, Abf, N, DH);

  // ---- layer 2: 256 -> 128, linear out ----
  launch_gemm(Abf, Wt2, hw, N, DH, DOUT, stream);
  hipMemsetAsync(agg, 0, (size_t)N * DOUT * 4, stream);
  k_agg<<<aggBlocks, TB, 0, stream>>>(rowp, colp, dinv, hw, agg, E, DOUT);
  k_fin_f32<<<(int)(((long long)N * DOUT + TB - 1) / TB), TB, 0, stream>>>(agg, hw, dinv, b2, (float*)d_out, N, DOUT);
}
